// Branch_4887672782920
// MI455X (gfx1250) — matrix operand order repaired and checked by reading, not compiled, not run on hardware
//
#include <hip/hip_runtime.h>
#include <hip/hip_fp16.h>

typedef __attribute__((ext_vector_type(16))) _Float16 v16h;
typedef __attribute__((ext_vector_type(8)))  _Float16 v8h;
typedef __attribute__((ext_vector_type(8)))  float    v8f;

#define TPB 256

// ---------------------------------------------------------------------------
// Generic batched TN GEMM: C[M,N] = A[M,K] * W[N,K]^T, f16 in, f32 out.
// One wave computes a 16x32 C macro-tile (two v_wmma_f32_16x16x32_f16 per
// K-step) so each A fragment is reused for two WMMAs.
// Per-lane fragment addressing per CDNA5 ISA 7.12.2 layout tables.
// Requires: M % 16 == 0, N % 32 == 0, K % 32 == 0.
// ---------------------------------------------------------------------------
__global__ __launch_bounds__(TPB) void k_gemm_tn(
    const _Float16* __restrict__ A, const _Float16* __restrict__ W,
    float* __restrict__ C, int M, int N, int K,
    long aBatch, long wBatch, long cBatch)
{
  const _Float16* Ab = A + (long)blockIdx.z * aBatch;
  const _Float16* Wb = W + (long)blockIdx.z * wBatch;
  float*          Cb = C + (long)blockIdx.z * cBatch;

  int wave    = blockIdx.x * (TPB / 32) + (threadIdx.x >> 5);
  int lane    = threadIdx.x & 31;
  int tilesN2 = N >> 5;
  if (wave >= (M >> 4) * tilesN2) return;      // wave-uniform: EXEC stays all-1s
  int tm  = (wave / tilesN2) << 4;
  int tn  = (wave % tilesN2) << 5;
  int r   = lane & 15;
  int khi = lane >> 4;

  const _Float16* aRow  = Ab + (long)(tm + r) * K + 8  * khi;  // A frag: k = kk+8khi+e(0..7), kk+16+8khi+e
  const _Float16* wRow0 = Wb + (long)(tn + r) * K + 8  * khi;  // B frag: k = kk+8khi+e(0..7), kk+16+8khi+e
  const _Float16* wRow1 = wRow0 + (long)16 * K;                // second 16 columns of the macro-tile

  v8f acc0 = {};
  v8f acc1 = {};
  for (int kk = 0; kk < K; kk += 32) {
    v8h a0  = *(const v8h*)(aRow  + kk);
    v8h a1  = *(const v8h*)(aRow  + kk + 16);
    v8h b00 = *(const v8h*)(wRow0 + kk);
    v8h b01 = *(const v8h*)(wRow0 + kk + 16);
    v8h b10 = *(const v8h*)(wRow1 + kk);
    v8h b11 = *(const v8h*)(wRow1 + kk + 16);
    v16h av  = __builtin_shufflevector(a0,  a1,  0,1,2,3,4,5,6,7,8,9,10,11,12,13,14,15);
    v16h bv0 = __builtin_shufflevector(b00, b01, 0,1,2,3,4,5,6,7,8,9,10,11,12,13,14,15);
    v16h bv1 = __builtin_shufflevector(b10, b11, 0,1,2,3,4,5,6,7,8,9,10,11,12,13,14,15);
    acc0 = __builtin_amdgcn_wmma_f32_16x16x32_f16(false, av, false, bv0, (short)0, acc0, false, false);
    acc1 = __builtin_amdgcn_wmma_f32_16x16x32_f16(false, av, false, bv1, (short)0, acc1, false, false);
  }
  float* cp = Cb + (long)(tm + 8 * khi) * N + tn + r;   // C: lane n=l&15, rows m=8*(l>>4)+v
  #pragma unroll
  for (int v = 0; v < 8; ++v) {
    cp[(long)v * N]      = acc0[v];
    cp[(long)v * N + 16] = acc1[v];
  }
}

// First layer: K=7 (not WMMA-friendly). h1[m][c] = sum_d x[m][d]*w1[c][d]
__global__ void k_lin7(const float* __restrict__ X, const float* __restrict__ W,
                       float* __restrict__ Y, long M)
{
  long i = blockIdx.x * (long)TPB + threadIdx.x;
  if (i >= M * 64) return;
  long m = i >> 6; int c = (int)(i & 63);
  const float* xp = X + m * 7;
  const float* wp = W + c * 7;
  float s = 0.f;
  #pragma unroll
  for (int d = 0; d < 7; ++d) s += xp[d] * wp[d];
  Y[i] = s;
}

// Column mean/var over (R,C) row-major. One block per column.
__global__ void k_colstats(const float* __restrict__ X, long R, int C,
                           float* __restrict__ mean, float* __restrict__ var)
{
  int c = blockIdx.x;
  float s = 0.f, s2 = 0.f;
  for (long r = threadIdx.x; r < R; r += TPB) {
    float v = X[r * C + c];
    s += v; s2 += v * v;
  }
  __shared__ float sh[TPB], sh2[TPB];
  sh[threadIdx.x] = s; sh2[threadIdx.x] = s2;
  __syncthreads();
  for (int o = TPB >> 1; o > 0; o >>= 1) {
    if ((int)threadIdx.x < o) { sh[threadIdx.x] += sh[threadIdx.x + o]; sh2[threadIdx.x] += sh2[threadIdx.x + o]; }
    __syncthreads();
  }
  if (threadIdx.x == 0) {
    float mu = sh[0] / (float)R;
    mean[c] = mu;
    var[c]  = sh2[0] / (float)R - mu * mu;
  }
}

// BN epilogue: y=(x-mu)*rsqrt(var+eps)*g+b ; act(0 none,1 relu,2 leaky0.2);
// optional residual add (after act), optional f32 out, up to two strided f16 outs.
__global__ void k_bn_ep(const float* __restrict__ Y, long total, int C,
                        const float* __restrict__ mean, const float* __restrict__ var,
                        const float* __restrict__ g, const float* __restrict__ bt,
                        int act, const float* __restrict__ res,
                        float* __restrict__ out32,
                        _Float16* __restrict__ o16a, int s16a, int off16a,
                        _Float16* __restrict__ o16b, int s16b, int off16b)
{
  long i = blockIdx.x * (long)TPB + threadIdx.x;
  if (i >= total) return;
  long r = i / C; int c = (int)(i - r * C);
  float v = Y[i];
  v = (v - mean[c]) * rsqrtf(var[c] + 1e-5f) * g[c] + bt[c];
  if (act == 1)      v = fmaxf(v, 0.f);
  else if (act == 2) v = (v > 0.f) ? v : 0.2f * v;
  if (res)   v += res[i];
  if (out32) out32[i] = v;
  if (o16a)  o16a[r * (long)s16a + off16a + c] = (_Float16)v;
  if (o16b)  o16b[r * (long)s16b + off16b + c] = (_Float16)v;
}

// f32 -> f16 copy with optional modular row bias and strided destination.
__global__ void k_cvt16(const float* __restrict__ X, long total, int C,
                        const float* __restrict__ rowBias, int biasMod,
                        _Float16* __restrict__ out, int oStride, int oOff)
{
  long i = blockIdx.x * (long)TPB + threadIdx.x;
  if (i >= total) return;
  long r = i / C; int c = (int)(i - r * C);
  float v = X[i];
  if (rowBias) v += rowBias[r % biasMod];
  out[r * (long)oStride + oOff + c] = (_Float16)v;
}

// Farthest point sampling: one block per batch; sequential S iterations.
__global__ void k_fps(const float* __restrict__ xyz, int ld, int N, int S,
                      int* __restrict__ outIdx)
{
  extern __shared__ char dynsm[];
  float* dist = (float*)dynsm;          // N
  float* rv   = dist + N;               // TPB
  int*   ri   = (int*)(rv + TPB);       // TPB
  int b = blockIdx.x;
  const float* P = xyz + (long)b * N * ld;
  for (int i = threadIdx.x; i < N; i += TPB) dist[i] = 1e10f;
  __syncthreads();
  int far = 0;
  for (int s = 0; s < S; ++s) {
    if (threadIdx.x == 0) outIdx[b * S + s] = far;
    float cx = P[(long)far * ld], cy = P[(long)far * ld + 1], cz = P[(long)far * ld + 2];
    float best = -1.f; int bi = 0;
    for (int i = threadIdx.x; i < N; i += TPB) {
      float dx = P[(long)i * ld] - cx, dy = P[(long)i * ld + 1] - cy, dz = P[(long)i * ld + 2] - cz;
      float nd = fminf(dist[i], dx * dx + dy * dy + dz * dz);
      dist[i] = nd;
      if (nd > best) { best = nd; bi = i; }
    }
    rv[threadIdx.x] = best; ri[threadIdx.x] = bi;
    __syncthreads();
    for (int o = TPB >> 1; o > 0; o >>= 1) {
      if ((int)threadIdx.x < o) {
        float ov = rv[threadIdx.x + o]; int oi = ri[threadIdx.x + o];
        if (ov > rv[threadIdx.x] || (ov == rv[threadIdx.x] && oi < ri[threadIdx.x])) {
          rv[threadIdx.x] = ov; ri[threadIdx.x] = oi;
        }
      }
      __syncthreads();
    }
    far = ri[0];
    __syncthreads();
  }
}

// Gather xyz rows: out[b,s,:3] = src[b, idx[b,s], :3]
__global__ void k_gather3(const float* __restrict__ src, int ld, int N,
                          const int* __restrict__ idx, int S,
                          float* __restrict__ out, long total)
{
  long i = blockIdx.x * (long)TPB + threadIdx.x;
  if (i >= total) return;
  long t = i; int d = (int)(t % 3); t /= 3; int s = (int)(t % S); int b = (int)(t / S);
  out[i] = src[((long)b * N + idx[b * S + s]) * ld + d];
}

// kNN: one block per (b,s); iterative 32x min-extraction.
__global__ void k_knn(const float* __restrict__ pts, int ld, int N,
                      const float* __restrict__ cent, int S, int* __restrict__ outIdx)
{
  extern __shared__ char dynsm[];
  float* d  = (float*)dynsm;           // N
  float* rv = d + N;                   // TPB
  int*   ri = (int*)(rv + TPB);        // TPB
  int bs = blockIdx.x; int b = bs / S;
  const float* P = pts + (long)b * N * ld;
  float cx = cent[bs * 3], cy = cent[bs * 3 + 1], cz = cent[bs * 3 + 2];
  for (int i = threadIdx.x; i < N; i += TPB) {
    float dx = P[(long)i * ld] - cx, dy = P[(long)i * ld + 1] - cy, dz = P[(long)i * ld + 2] - cz;
    d[i] = dx * dx + dy * dy + dz * dz;
  }
  __syncthreads();
  for (int k = 0; k < 32; ++k) {
    float best = 1e30f; int bi = 0;
    for (int i = threadIdx.x; i < N; i += TPB)
      if (d[i] < best) { best = d[i]; bi = i; }
    rv[threadIdx.x] = best; ri[threadIdx.x] = bi;
    __syncthreads();
    for (int o = TPB >> 1; o > 0; o >>= 1) {
      if ((int)threadIdx.x < o) {
        float ov = rv[threadIdx.x + o]; int oi = ri[threadIdx.x + o];
        if (ov < rv[threadIdx.x] || (ov == rv[threadIdx.x] && oi < ri[threadIdx.x])) {
          rv[threadIdx.x] = ov; ri[threadIdx.x] = oi;
        }
      }
      __syncthreads();
    }
    if (threadIdx.x == 0) { outIdx[(long)bs * 32 + k] = ri[0]; d[ri[0]] = 1e30f; }
    __syncthreads();
  }
}

// Build grouped features (f16): [grouped - center, center] for each (b,s,j).
__global__ void k_group(const float* __restrict__ F, int N, int C,
                        const int* __restrict__ fpsIdx, const int* __restrict__ knnIdx,
                        int S, _Float16* __restrict__ out, long total)
{
  long i = blockIdx.x * (long)TPB + threadIdx.x;
  if (i >= total) return;
  int twoC = 2 * C;
  long row = i / twoC; int c = (int)(i - row * twoC);
  int  j  = (int)(row & 31);
  long bs = row >> 5;
  int  s  = (int)(bs % S); int b = (int)(bs / S);
  const float* Fb = F + (long)b * N * C;
  float center = Fb[(long)fpsIdx[b * S + s] * C + (c & (C - 1))];
  float v;
  if (c < C) v = Fb[(long)knnIdx[bs * 32 + j] * C + c] - center;
  else       v = center;
  out[i] = (_Float16)v;
}

// Max over 32 samples per group: out[r][c] = max_j Y[r*32+j][c]
__global__ void k_maxpool32(const float* __restrict__ Y, long R, int C,
                            float* __restrict__ out)
{
  long i = blockIdx.x * (long)TPB + threadIdx.x;
  if (i >= R * C) return;
  long r = i / C; int c = (int)(i - r * C);
  const float* p = Y + (r * 32) * (long)C + c;
  float m = p[0];
  for (int j = 1; j < 32; ++j) m = fmaxf(m, p[(long)j * C]);
  out[i] = m;
}

// Row softmax of e[b,n,:], written transposed: AT[b,m,n]. One block per (b,n).
__global__ void k_softmaxT(const float* __restrict__ E, float* __restrict__ AT)
{
  int bn = blockIdx.x; int b = bn >> 8; int n = bn & 255;
  int t = threadIdx.x;
  float v = E[(long)bn * 256 + t];
  __shared__ float sh[TPB];
  sh[t] = v; __syncthreads();
  for (int o = 128; o > 0; o >>= 1) { if (t < o) sh[t] = fmaxf(sh[t], sh[t + o]); __syncthreads(); }
  float mx = sh[0]; __syncthreads();
  float e = __expf(v - mx);
  sh[t] = e; __syncthreads();
  for (int o = 128; o > 0; o >>= 1) { if (t < o) sh[t] += sh[t + o]; __syncthreads(); }
  AT[((long)b * 256 + t) * 256 + n] = e / sh[0];
}

// Normalize AT rows by (1e-9 + row sum) -> f16. One block per (b,m).
__global__ void k_colnormT(const float* __restrict__ AT, _Float16* __restrict__ AT16)
{
  int bm = blockIdx.x; int t = threadIdx.x;
  float v = AT[(long)bm * 256 + t];
  __shared__ float sh[TPB];
  sh[t] = v; __syncthreads();
  for (int o = 128; o > 0; o >>= 1) { if (t < o) sh[t] += sh[t + o]; __syncthreads(); }
  AT16[(long)bm * 256 + t] = (_Float16)(v / (1e-9f + sh[0]));
}

// tT16[b,n,c] = x32[b,n,c] - xr32[b,c,n]
__global__ void k_subT(const float* __restrict__ X, const float* __restrict__ XR,
                       _Float16* __restrict__ T16, long total)
{
  long i = blockIdx.x * (long)TPB + threadIdx.x;
  if (i >= total) return;
  long b = i >> 16; int nc = (int)(i & 65535); int n = nc >> 8; int c = nc & 255;
  T16[i] = (_Float16)(X[i] - XR[(b << 16) + ((long)c << 8) + n]);
}

// out[b,c] = max_n Y[b*256+n][c], Y (B*256, 1024)
__global__ void k_finalmax(const float* __restrict__ Y, float* __restrict__ out)
{
  long i = blockIdx.x * (long)TPB + threadIdx.x;
  if (i >= 32 * 1024) return;
  int b = (int)(i >> 10); int c = (int)(i & 1023);
  const float* p = Y + (long)b * 256 * 1024 + c;
  float m = -1e30f;
  for (int n = 0; n < 256; ++n) m = fmaxf(m, p[(long)n * 1024]);
  out[i] = m;
}

// ---------------------------------------------------------------------------
extern "C" void kernel_launch(void* const* d_in, const int* in_sizes, int n_in,
                              void* d_out, int out_size, void* d_ws, size_t ws_size,
                              hipStream_t stream)
{
  (void)in_sizes; (void)n_in; (void)out_size; (void)ws_size;
  const int B = 32;

  const float* x    = (const float*)d_in[0];
  const float* w1   = (const float*)d_in[1];
  const float* g1   = (const float*)d_in[2];
  const float* b1   = (const float*)d_in[3];
  const float* w2   = (const float*)d_in[4];
  const float* g2   = (const float*)d_in[5];
  const float* b2   = (const float*)d_in[6];
  const float* l0w1 = (const float*)d_in[7];
  const float* l0g1 = (const float*)d_in[8];
  const float* l0b1 = (const float*)d_in[9];
  const float* l0w2 = (const float*)d_in[10];
  const float* l0g2 = (const float*)d_in[11];
  const float* l0b2 = (const float*)d_in[12];
  const float* l1w1 = (const float*)d_in[13];
  const float* l1g1 = (const float*)d_in[14];
  const float* l1b1 = (const float*)d_in[15];
  const float* l1w2 = (const float*)d_in[16];
  const float* l1g2 = (const float*)d_in[17];
  const float* l1b2 = (const float*)d_in[18];
  const float* pw1  = (const float*)d_in[19];
  const float* pg1  = (const float*)d_in[20];
  const float* pb1  = (const float*)d_in[21];
  const float* pw2  = (const float*)d_in[22];
  const float* pg2  = (const float*)d_in[23];
  const float* pb2  = (const float*)d_in[24];
  // sa[l]: 25+7l = wqk, wv, bv, wt, bt(unused: absorbed by BN mean), g, b
  const float* fw = (const float*)d_in[53];
  const float* fg = (const float*)d_in[54];
  const float* fb = (const float*)d_in[55];

  // ---- workspace carve ----
  char* ws = (char*)d_ws;
  size_t off = 0;
  auto alloc = [&](size_t bytes) -> char* {
    char* p = ws + off;
    off += (bytes + 255) & ~(size_t)255;
    return p;
  };
  float*    BIGA  = (float*)alloc((size_t)67108864 * 4);   // 256 MiB f32
  _Float16* BIGB  = (_Float16*)alloc((size_t)67108864 * 2);// 128 MiB f16
  float*    pts32 = (float*)alloc((size_t)65536 * 64 * 4);
  float*    f0    = (float*)alloc((size_t)16384 * 128 * 4);
  float*    f1    = (float*)alloc((size_t)8192 * 256 * 4);
  int*      fi1   = (int*)alloc((size_t)B * 512 * 4);
  float*    nx1   = (float*)alloc((size_t)B * 512 * 3 * 4);
  int*      knn1  = (int*)alloc((size_t)B * 512 * 32 * 4);
  int*      fi2   = (int*)alloc((size_t)B * 256 * 4);
  float*    nx2   = (float*)alloc((size_t)B * 256 * 3 * 4);
  int*      knn2  = (int*)alloc((size_t)B * 256 * 32 * 4);
  _Float16* f1c16 = (_Float16*)alloc((size_t)8192 * 256 * 2);
  float*    mean  = (float*)alloc(1024 * 4);
  float*    var   = (float*)alloc(1024 * 4);
  _Float16* w2h   = (_Float16*)alloc((size_t)64 * 64 * 2);
  _Float16* l0w1h = (_Float16*)alloc((size_t)128 * 128 * 2);
  _Float16* l0w2h = (_Float16*)alloc((size_t)128 * 128 * 2);
  _Float16* l1w1h = (_Float16*)alloc((size_t)256 * 256 * 2);
  _Float16* l1w2h = (_Float16*)alloc((size_t)256 * 256 * 2);
  _Float16* pw1h  = (_Float16*)alloc((size_t)256 * 256 * 2);
  _Float16* pw2h  = (_Float16*)alloc((size_t)256 * 256 * 2);
  _Float16* wqkh[4]; _Float16* wvh[4]; _Float16* wth[4];
  for (int l = 0; l < 4; ++l) {
    wqkh[l] = (_Float16*)alloc((size_t)64 * 256 * 2);
    wvh[l]  = (_Float16*)alloc((size_t)256 * 256 * 2);
    wth[l]  = (_Float16*)alloc((size_t)256 * 256 * 2);
  }
  _Float16* fwh = (_Float16*)alloc((size_t)1024 * 1280 * 2);

  // SA-phase overlays inside BIGA/BIGB (the l0/l1 big buffers are dead by then)
  char* sab = (char*)BIGA;
  float*    x32  = (float*)sab;      sab += (size_t)8192 * 256 * 4;
  _Float16* xT16 = (_Float16*)sab;   sab += (size_t)8192 * 256 * 2;
  float*    q32  = (float*)sab;      sab += (size_t)8192 * 64 * 4;
  _Float16* q16  = (_Float16*)sab;   sab += (size_t)8192 * 64 * 2;
  float*    e32  = (float*)sab;      sab += (size_t)8192 * 256 * 4;
  float*    aT32 = (float*)sab;      sab += (size_t)8192 * 256 * 4;
  _Float16* aT16 = (_Float16*)sab;   sab += (size_t)8192 * 256 * 2;
  float*    v32  = (float*)sab;      sab += (size_t)8192 * 256 * 4;
  _Float16* v16  = (_Float16*)sab;   sab += (size_t)8192 * 256 * 2;
  float*    xr32 = (float*)sab;      sab += (size_t)8192 * 256 * 4;
  _Float16* tT16 = (_Float16*)sab;   sab += (size_t)8192 * 256 * 2;
  float*    y32  = (float*)sab;      sab += (size_t)8192 * 256 * 4;
  float*    fin32 = (float*)sab;     sab += (size_t)8192 * 1024 * 4;
  _Float16* z16  = (_Float16*)BIGB;  // 8192 x 1280 f16

  auto cdiv = [](long a, long b) -> unsigned { return (unsigned)((a + b - 1) / b); };
  auto gemm = [&](const _Float16* A, const _Float16* W, float* C_, int M, int N, int K,
                  int nb, long aB, long wB, long cB) {
    long tiles = (long)(M >> 4) * (N >> 5);   // 16x32 macro-tiles per wave
    k_gemm_tn<<<dim3(cdiv(tiles, TPB / 32), 1, nb), dim3(TPB), 0, stream>>>(A, W, C_, M, N, K, aB, wB, cB);
  };
  auto stats = [&](const float* Y, long R, int C) {
    k_colstats<<<dim3(C), dim3(TPB), 0, stream>>>(Y, R, C, mean, var);
  };
  auto cvtW = [&](const float* src, int R, int C, _Float16* dst) {
    long tot = (long)R * C;
    k_cvt16<<<dim3(cdiv(tot, TPB)), dim3(TPB), 0, stream>>>(src, tot, C, (const float*)nullptr, 1, dst, C, 0);
  };
  _Float16* const H16N = nullptr;
  float* const F32N = nullptr;

  // ---- weight fp16 conversions ----
  cvtW(w2, 64, 64, w2h);
  cvtW(l0w1, 128, 128, l0w1h); cvtW(l0w2, 128, 128, l0w2h);
  cvtW(l1w1, 256, 256, l1w1h); cvtW(l1w2, 256, 256, l1w2h);
  cvtW(pw1, 256, 256, pw1h);   cvtW(pw2, 256, 256, pw2h);
  for (int l = 0; l < 4; ++l) {
    cvtW((const float*)d_in[25 + 7 * l + 0], 64, 256, wqkh[l]);
    cvtW((const float*)d_in[25 + 7 * l + 1], 256, 256, wvh[l]);
    cvtW((const float*)d_in[25 + 7 * l + 3], 256, 256, wth[l]);
  }
  cvtW(fw, 1024, 1280, fwh);

  // ---- stage 1: 7->64->64 pointwise MLP with BN+ReLU ----
  long M1 = (long)B * 2048;
  k_lin7<<<dim3(cdiv(M1 * 64, TPB)), dim3(TPB), 0, stream>>>(x, w1, BIGA, M1);
  stats(BIGA, M1, 64);
  k_bn_ep<<<dim3(cdiv(M1 * 64, TPB)), dim3(TPB), 0, stream>>>(BIGA, M1 * 64, 64, mean, var, g1, b1, 1,
      F32N, F32N, BIGB, 64, 0, H16N, 0, 0);
  gemm(BIGB, w2h, BIGA, (int)M1, 64, 64, 1, 0, 0, 0);
  stats(BIGA, M1, 64);
  k_bn_ep<<<dim3(cdiv(M1 * 64, TPB)), dim3(TPB), 0, stream>>>(BIGA, M1 * 64, 64, mean, var, g2, b2, 1,
      F32N, pts32, H16N, 0, 0, H16N, 0, 0);

  // ---- sample_and_group(512, 32) on raw xyz (stride 7) ----
  k_fps<<<dim3(B), dim3(TPB), 2048 * 4 + TPB * 8, stream>>>(x, 7, 2048, 512, fi1);
  k_gather3<<<dim3(cdiv((long)B * 512 * 3, TPB)), dim3(TPB), 0, stream>>>(x, 7, 2048, fi1, 512, nx1, (long)B * 512 * 3);
  k_knn<<<dim3(B * 512), dim3(TPB), 2048 * 4 + TPB * 8, stream>>>(x, 7, 2048, nx1, 512, knn1);
  long R0 = (long)B * 512 * 32;
  k_group<<<dim3(cdiv(R0 * 128, TPB)), dim3(TPB), 0, stream>>>(pts32, 2048, 64, fi1, knn1, 512, BIGB, R0 * 128);

  // ---- local_op l0: 128->128->128, BN+ReLU, maxpool over 32 ----
  gemm(BIGB, l0w1h, BIGA, (int)R0, 128, 128, 1, 0, 0, 0);
  stats(BIGA, R0, 128);
  k_bn_ep<<<dim3(cdiv(R0 * 128, TPB)), dim3(TPB), 0, stream>>>(BIGA, R0 * 128, 128, mean, var, l0g1, l0b1, 1,
      F32N, F32N, BIGB, 128, 0, H16N, 0, 0);
  gemm(BIGB, l0w2h, BIGA, (int)R0, 128, 128, 1, 0, 0, 0);
  stats(BIGA, R0, 128);
  k_bn_ep<<<dim3(cdiv(R0 * 128, TPB)), dim3(TPB), 0, stream>>>(BIGA, R0 * 128, 128, mean, var, l0g2, l0b2, 1,
      F32N, BIGA, H16N, 0, 0, H16N, 0, 0);
  k_maxpool32<<<dim3(cdiv((long)B * 512 * 128, TPB)), dim3(TPB), 0, stream>>>(BIGA, (long)B * 512, 128, f0);

  // ---- sample_and_group(256, 32) on nx1 (stride 3), feats f0 ----
  k_fps<<<dim3(B), dim3(TPB), 512 * 4 + TPB * 8, stream>>>(nx1, 3, 512, 256, fi2);
  k_gather3<<<dim3(cdiv((long)B * 256 * 3, TPB)), dim3(TPB), 0, stream>>>(nx1, 3, 512, fi2, 256, nx2, (long)B * 256 * 3);
  k_knn<<<dim3(B * 256), dim3(TPB), 512 * 4 + TPB * 8, stream>>>(nx1, 3, 512, nx2, 256, knn2);
  long R1 = (long)B * 256 * 32;
  k_group<<<dim3(cdiv(R1 * 256, TPB)), dim3(TPB), 0, stream>>>(f0, 512, 128, fi2, knn2, 256, BIGB, R1 * 256);

  // ---- local_op l1: 256->256->256 ----
  gemm(BIGB, l1w1h, BIGA, (int)R1, 256, 256, 1, 0, 0, 0);
  stats(BIGA, R1, 256);
  k_bn_ep<<<dim3(cdiv(R1 * 256, TPB)), dim3(TPB), 0, stream>>>(BIGA, R1 * 256, 256, mean, var, l1g1, l1b1, 1,
      F32N, F32N, BIGB, 256, 0, H16N, 0, 0);
  gemm(BIGB, l1w2h, BIGA, (int)R1, 256, 256, 1, 0, 0, 0);
  stats(BIGA, R1, 256);
  k_bn_ep<<<dim3(cdiv(R1 * 256, TPB)), dim3(TPB), 0, stream>>>(BIGA, R1 * 256, 256, mean, var, l1g2, l1b2, 1,
      F32N, BIGA, H16N, 0, 0, H16N, 0, 0);
  k_maxpool32<<<dim3(cdiv((long)B * 256 * 256, TPB)), dim3(TPB), 0, stream>>>(BIGA, (long)B * 256, 256, f1);

  long R2 = (long)B * 256;               // 8192 rows of 256 channels
  long E2 = R2 * 256;                    // 2,097,152 elements
  // f1 -> f16 (GEMM input) and into z columns [1024,1280)
  k_cvt16<<<dim3(cdiv(E2, TPB)), dim3(TPB), 0, stream>>>(f1, E2, 256, (const float*)nullptr, 1, f1c16, 256, 0);
  k_cvt16<<<dim3(cdiv(E2, TPB)), dim3(TPB), 0, stream>>>(f1, E2, 256, (const float*)nullptr, 1, z16, 1280, 1024);

  // ---- pw1 / pw2 ----
  gemm(f1c16, pw1h, y32, (int)R2, 256, 256, 1, 0, 0, 0);
  stats(y32, R2, 256);
  k_bn_ep<<<dim3(cdiv(E2, TPB)), dim3(TPB), 0, stream>>>(y32, E2, 256, mean, var, pg1, pb1, 1,
      F32N, F32N, tT16, 256, 0, H16N, 0, 0);
  gemm(tT16, pw2h, y32, (int)R2, 256, 256, 1, 0, 0, 0);
  stats(y32, R2, 256);
  k_bn_ep<<<dim3(cdiv(E2, TPB)), dim3(TPB), 0, stream>>>(y32, E2, 256, mean, var, pg2, pb2, 1,
      F32N, x32, xT16, 256, 0, H16N, 0, 0);

  // ---- 4 self-attention layers ----
  for (int l = 0; l < 4; ++l) {
    const float* bv = (const float*)d_in[25 + 7 * l + 2];
    const float* sg = (const float*)d_in[25 + 7 * l + 5];
    const float* sb = (const float*)d_in[25 + 7 * l + 6];
    // Q = xT @ wqk^T  (8192 x 64); k is Q transposed, so e = Q Q^T per batch
    gemm(xT16, wqkh[l], q32, (int)R2, 64, 256, 1, 0, 0, 0);
    k_cvt16<<<dim3(cdiv(R2 * 64, TPB)), dim3(TPB), 0, stream>>>(q32, R2 * 64, 64, (const float*)nullptr, 1, q16, 64, 0);
    gemm(q16, q16, e32, 256, 256, 64, B, 256 * 64, 256 * 64, 256 * 256);
    k_softmaxT<<<dim3(B * 256), dim3(TPB), 0, stream>>>(e32, aT32);
    k_colnormT<<<dim3(B * 256), dim3(TPB), 0, stream>>>(aT32, aT16);
    // v[d,n] = wv @ x  (per batch; A=wv shared via batch stride 0), + bv, -> f16
    gemm(wvh[l], xT16, v32, 256, 256, 256, B, 0, 256 * 256, 256 * 256);
    k_cvt16<<<dim3(cdiv(E2, TPB)), dim3(TPB), 0, stream>>>(v32, E2, 256, bv, 256, v16, 256, 0);
    // xr[d,m] = v @ a  (A=v (d,n), W=aT (m,n))
    gemm(v16, aT16, xr32, 256, 256, 256, B, 256 * 256, 256 * 256, 256 * 256);
    // t = x - xr (transposed read), then xr2 = t @ wt^T (bt absorbed by BN)
    k_subT<<<dim3(cdiv(E2, TPB)), dim3(TPB), 0, stream>>>(x32, xr32, tT16, E2);
    gemm(tT16, wth[l], y32, (int)R2, 256, 256, 1, 0, 0, 0);
    stats(y32, R2, 256);
    // x = x + relu(bn(xr2)); emit new xT16 and z slice [256*l, 256*l+256)
    k_bn_ep<<<dim3(cdiv(E2, TPB)), dim3(TPB), 0, stream>>>(y32, E2, 256, mean, var, sg, sb, 1,
        x32, x32, xT16, 256, 0, z16, 1280, 256 * l);
  }

  // ---- final 1280 -> 1024, BN + leaky(0.2), max over points ----
  gemm(z16, fwh, fin32, (int)R2, 1024, 1280, 1, 0, 0, 0);
  stats(fin32, R2, 1024);
  k_bn_ep<<<dim3(cdiv(R2 * 1024, TPB)), dim3(TPB), 0, stream>>>(fin32, R2 * 1024, 1024, mean, var, fg, fb, 2,
      F32N, fin32, H16N, 0, 0, H16N, 0, 0);
  k_finalmax<<<dim3(cdiv((long)32 * 1024, TPB)), dim3(TPB), 0, stream>>>(fin32, (float*)d_out);
}
